// GGNNContext_14336600834164
// MI455X (gfx1250) — matrix operand order repaired and checked by reading, not compiled, not run on hardware
//
#include <hip/hip_runtime.h>
#include <hip/hip_bf16.h>
#include <math.h>

// ---------------------------------------------------------------------------
// GBNet GGNN forward for MI455X (gfx1250, wave32).
// Dense GEMMs on the matrix pipes via v_wmma_f32_16x16x32_bf16:
// fp32 -> bf16 pair-packed into double-buffered LDS, fp32 accumulate.
// Tile 128x128x32, 256 threads = 8 waves, each wave 32x64 (2x4 WMMA tiles).
// Pipeline: store tile k to LDS[buf], issue global loads for tile k+1,
// one barrier, then 8x WMMA on LDS[buf]  (global latency hidden under WMMA).
// ---------------------------------------------------------------------------

#define H     1024
#define Q     256
#define HMID  512
#define DEMB  300
#define NE    151
#define NP    51
#define NEDGE 3
#define NOBJ  2048
#define NREL  16384
#define TSTEP 3
#define CAT_O (7*Q)   /* 1792 */
#define CAT_I (3*Q)   /* 768  */

#define FLAG_RELU 1
#define FLAG_ACC  2

#define SROW 20      /* LDS row stride in dwords (16B-aligned rows, bank spread) */

typedef __attribute__((ext_vector_type(16))) __bf16 v16bf;
typedef __attribute__((ext_vector_type(8)))  float  v8f;

static __device__ __forceinline__ unsigned short f2bf(float x) {
    union { float f; unsigned u; } v; v.f = x;
    unsigned r = v.u + 0x7FFFu + ((v.u >> 16) & 1u);   // round-to-nearest-even
    return (unsigned short)(r >> 16);
}
static __device__ __forceinline__ unsigned pack2bf(float lo, float hi) {
    return (unsigned)f2bf(lo) | ((unsigned)f2bf(hi) << 16);
}
static __device__ __forceinline__ float sigmf(float x) { return 1.0f / (1.0f + expf(-x)); }
static __device__ __forceinline__ int imin(int a, int b) { return a < b ? a : b; }

// ---------------------------------------------------------------------------
// Tiled WMMA GEMM:  C[M,N] = op(A)[M,K] @ op(B)[K,N] (+bias) (+C) (relu?)
//   TA=false: A row-major [M,K].  TA=true: A stored [K,M] (A^T used).
//   TB=false: B row-major [K,N].  TB=true: B stored [N,K] (B^T used).
// ---------------------------------------------------------------------------
template <bool TA, bool TB>
__global__ __launch_bounds__(256) void gemm_k(const float* __restrict__ A,
                                              const float* __restrict__ B,
                                              const float* __restrict__ bias,
                                              float* __restrict__ C,
                                              int M, int N, int K,
                                              int lda, int ldb, int ldc, int flags) {
    __shared__ unsigned sA32[2][128 * SROW];   // [buf][m][k/2] packed bf16 pairs
    __shared__ unsigned sB32[2][128 * SROW];   // [buf][n][k/2] packed bf16 pairs (N-major)

    const int tid  = threadIdx.x;
    const int wave = tid >> 5;
    const int lane = tid & 31;
    const int hi   = lane >> 4;       // 0: lanes 0-15, 1: lanes 16-31
    const int l16  = lane & 15;
    const int wm   = (wave >> 1) * 32;   // wave row offset (4 wave-rows)
    const int wn   = (wave & 1) * 64;    // wave col offset (2 wave-cols)
    const int bm   = blockIdx.y * 128;
    const int bn   = blockIdx.x * 128;

    const bool fullMN = (bm + 128 <= M) && (bn + 128 <= N);

    v8f acc[2][4] = {};
    unsigned pa[8], pb[8];             // prefetched, packed bf16 pairs

    // ---- load one 128x32 A-tile + 32x128 B-tile into registers ----
    auto load_regs = [&](int k0) {
        const bool fast = fullMN && (k0 + 32 <= K);
        if (fast) {
            #pragma unroll
            for (int j = 0; j < 8; ++j) {
                int p = j * 256 + tid;               // 2048 pairs per matrix
                if (!TA) {
                    int m = p >> 4, kp = (p & 15) * 2;
                    const float* a = A + (size_t)(bm + m) * lda + (k0 + kp);
                    pa[j] = pack2bf(a[0], a[1]);
                } else {
                    int m = p & 127, kp = (p >> 7) * 2;
                    const float* a = A + (size_t)(k0 + kp) * lda + (bm + m);
                    pa[j] = pack2bf(a[0], a[lda]);
                }
                if (!TB) {
                    int n = p & 127, kp = (p >> 7) * 2;
                    const float* b = B + (size_t)(k0 + kp) * ldb + (bn + n);
                    pb[j] = pack2bf(b[0], b[ldb]);
                } else {
                    int n = p >> 4, kp = (p & 15) * 2;
                    const float* b = B + (size_t)(bn + n) * ldb + (k0 + kp);
                    pb[j] = pack2bf(b[0], b[1]);
                }
            }
        } else {
            // clamp-index + select (loads always in-bounds, zeroed by cndmask)
            #pragma unroll
            for (int j = 0; j < 8; ++j) {
                int p = j * 256 + tid;
                {
                    int m, kp;
                    if (!TA) { m = p >> 4; kp = (p & 15) * 2; }
                    else     { m = p & 127; kp = (p >> 7) * 2; }
                    int gm = bm + m, gk = k0 + kp;
                    int cm = imin(gm, M - 1), ck0 = imin(gk, K - 1), ck1 = imin(gk + 1, K - 1);
                    float v0, v1;
                    if (!TA) {
                        const float* a = A + (size_t)cm * lda;
                        v0 = a[ck0]; v1 = a[ck1];
                    } else {
                        v0 = A[(size_t)ck0 * lda + cm];
                        v1 = A[(size_t)ck1 * lda + cm];
                    }
                    v0 = (gm < M && gk < K)     ? v0 : 0.0f;
                    v1 = (gm < M && gk + 1 < K) ? v1 : 0.0f;
                    pa[j] = pack2bf(v0, v1);
                }
                {
                    int n, kp;
                    if (!TB) { n = p & 127; kp = (p >> 7) * 2; }
                    else     { n = p >> 4;  kp = (p & 15) * 2; }
                    int gn = bn + n, gk = k0 + kp;
                    int cn = imin(gn, N - 1), ck0 = imin(gk, K - 1), ck1 = imin(gk + 1, K - 1);
                    float v0, v1;
                    if (!TB) {
                        v0 = B[(size_t)ck0 * ldb + cn];
                        v1 = B[(size_t)ck1 * ldb + cn];
                    } else {
                        const float* b = B + (size_t)cn * ldb;
                        v0 = b[ck0]; v1 = b[ck1];
                    }
                    v0 = (gn < N && gk < K)     ? v0 : 0.0f;
                    v1 = (gn < N && gk + 1 < K) ? v1 : 0.0f;
                    pb[j] = pack2bf(v0, v1);
                }
            }
        }
    };

    auto store_regs = [&](int buf) {
        #pragma unroll
        for (int j = 0; j < 8; ++j) {
            int p = j * 256 + tid;
            int m, ka;
            if (!TA) { m = p >> 4;  ka = p & 15;  }
            else     { m = p & 127; ka = p >> 7;  }
            sA32[buf][m * SROW + ka] = pa[j];
            int n, kb;
            if (!TB) { n = p & 127; kb = p >> 7;  }
            else     { n = p >> 4;  kb = p & 15;  }
            sB32[buf][n * SROW + kb] = pb[j];
        }
    };

    const int nk = (K + 31) >> 5;
    load_regs(0);

    for (int it = 0; it < nk; ++it) {
        const int buf = it & 1;
        store_regs(buf);
        if (it + 1 < nk) load_regs((it + 1) << 5);   // global loads fly under WMMA
        __syncthreads();   // single barrier per step (double-buffered LDS)

        // ---- A fragments (contiguous k: lanes 0-15: K 0..15, 16-31: K 16..31) ----
        v16bf afr[2];
        #pragma unroll
        for (int ti = 0; ti < 2; ++ti) {
            int m = wm + ti * 16 + l16;
            union { v16bf v; unsigned u[8]; } fa;
            #pragma unroll
            for (int p = 0; p < 8; ++p) {
                fa.u[p] = sA32[buf][m * SROW + hi * 8 + p];
            }
            afr[ti] = fa.v;
        }
        // ---- B fragments (lane = column; lanes 0-15: K 0..15, 16-31: K 16..31) ----
        v16bf bfr[4];
        #pragma unroll
        for (int tj = 0; tj < 4; ++tj) {
            int n = wn + tj * 16 + l16;
            union { v16bf v; unsigned u[8]; } fb;
            #pragma unroll
            for (int p = 0; p < 8; ++p)
                fb.u[p] = sB32[buf][n * SROW + hi * 8 + p];
            bfr[tj] = fb.v;
        }

        #pragma unroll
        for (int ti = 0; ti < 2; ++ti)
            #pragma unroll
            for (int tj = 0; tj < 4; ++tj)
                acc[ti][tj] = __builtin_amdgcn_wmma_f32_16x16x32_bf16(
                    false, afr[ti], false, bfr[tj], (short)0, acc[ti][tj], false, false);
    }

    // ---- epilogue: D layout row = r + 8*hi, col = l16 within each 16x16 ----
    #pragma unroll
    for (int ti = 0; ti < 2; ++ti)
        #pragma unroll
        for (int tj = 0; tj < 4; ++tj)
            #pragma unroll
            for (int r = 0; r < 8; ++r) {
                int row = bm + wm + ti * 16 + r + 8 * hi;
                int col = bn + wn + tj * 16 + l16;
                if (row < M && col < N) {
                    float v = acc[ti][tj][r];
                    if (bias) v += bias[col];
                    if (flags & FLAG_ACC)  v += C[(size_t)row * ldc + col];
                    if (flags & FLAG_RELU) v = fmaxf(v, 0.0f);
                    C[(size_t)row * ldc + col] = v;
                }
            }
}

// ---------------------------------------------------------------------------
// Elementwise / support kernels
// ---------------------------------------------------------------------------
__global__ void k_zero(float* p, size_t n) {
    size_t i = (size_t)blockIdx.x * blockDim.x + threadIdx.x;
    for (; i < n; i += (size_t)gridDim.x * blockDim.x) p[i] = 0.0f;
}

// img2ont_ent = softmax(one_hot(labels)) : hot -> e/(e+NE-1), else 1/(e+NE-1)
__global__ void k_obj_probs(const int* __restrict__ labels, float* __restrict__ out) {
    size_t i = (size_t)blockIdx.x * blockDim.x + threadIdx.x;
    size_t n = (size_t)NOBJ * NE;
    if (i >= n) return;
    int r = (int)(i / NE), c = (int)(i % NE);
    const float e1 = 2.718281828459045f;
    const float den = e1 + (float)(NE - 1);
    out[i] = (labels[r] == c) ? (e1 / den) : (1.0f / den);
}

__global__ void k_softmax_rows(const float* __restrict__ in, float* __restrict__ out,
                               int rows, int cols) {
    int r = blockIdx.x * blockDim.x + threadIdx.x;
    if (r >= rows) return;
    const float* x = in + (size_t)r * cols;
    float* y = out + (size_t)r * cols;
    float m = -3.4e38f;
    for (int c = 0; c < cols; ++c) m = fmaxf(m, x[c]);
    float s = 0.0f;
    for (int c = 0; c < cols; ++c) { float e = expf(x[c] - m); y[c] = e; s += e; }
    float inv = 1.0f / s;
    for (int c = 0; c < cols; ++c) y[c] *= inv;
}

// cat_ip[:, 0:256] = m_ie[sub], cat_ip[:, 256:512] = m_ie[obj]
__global__ void k_gather2(const float* __restrict__ mie, const int* __restrict__ sub,
                          const int* __restrict__ obj, float* __restrict__ cat) {
    size_t i = (size_t)blockIdx.x * blockDim.x + threadIdx.x;
    size_t n = (size_t)NREL * 512;
    if (i >= n) return;
    int r = (int)(i >> 9), j = (int)(i & 511);
    int src = (j < 256) ? sub[r] : obj[r];
    cat[(size_t)r * CAT_I + j] = mie[(size_t)src * Q + (j & 255)];
}

// scatter-add m_ip rows into cat_ie columns [colOff, colOff+256)
__global__ void k_segadd(const float* __restrict__ mip, const int* __restrict__ inds,
                         float* __restrict__ cat, int colOff) {
    size_t i = (size_t)blockIdx.x * blockDim.x + threadIdx.x;
    size_t n = (size_t)NREL * Q;
    if (i >= n) return;
    int r = (int)(i >> 8), q = (int)(i & 255);
    atomicAdd(&cat[(size_t)inds[r] * CAT_I + colOff + q], mip[i]);
}

// rp = sigmoid(rp) * h
__global__ void k_gru_rh(float* rp, const float* __restrict__ h, size_t n) {
    size_t i = (size_t)blockIdx.x * blockDim.x + threadIdx.x;
    if (i < n) rp[i] = sigmf(rp[i]) * h[i];
}

// h = (1-z)*h + z*tanh(hp),  z = sigmoid(zp)
__global__ void k_gru_out(const float* __restrict__ zp, const float* __restrict__ hp,
                          float* h, size_t n) {
    size_t i = (size_t)blockIdx.x * blockDim.x + threadIdx.x;
    if (i < n) {
        float z = sigmf(zp[i]);
        h[i] = (1.0f - z) * h[i] + z * tanhf(hp[i]);
    }
}

// ---------------------------------------------------------------------------
// Host side
// ---------------------------------------------------------------------------
struct Lin { const float* w; const float* b; };
struct Gru { Lin u3, u4, u5, w3, w4, w5; };

static void gemm(hipStream_t s, bool ta, bool tb,
                 const float* A, const float* B, const float* bias, float* C,
                 int M, int N, int K, int lda, int ldb, int ldc, int flags) {
    dim3 grid((N + 127) / 128, (M + 127) / 128), blk(256, 1, 1);
    // only 3 op-combinations are ever used; do not instantiate <true,true>
    if (ta)      gemm_k<true,  false><<<grid, blk, 0, s>>>(A, B, bias, C, M, N, K, lda, ldb, ldc, flags);
    else if (tb) gemm_k<false, true ><<<grid, blk, 0, s>>>(A, B, bias, C, M, N, K, lda, ldb, ldc, flags);
    else         gemm_k<false, false><<<grid, blk, 0, s>>>(A, B, bias, C, M, N, K, lda, ldb, ldc, flags);
}

static void zero(hipStream_t s, float* p, size_t n) {
    int blocks = (int)((n + 255) / 256); if (blocks > 65535) blocks = 65535;
    k_zero<<<blocks, 256, 0, s>>>(p, n);
}

// 2-layer MLP: out = act1(act0(x@l0.w+b)@l1.w+b)
static void mlp2(hipStream_t s, const float* x, int M, int d0, int d1, int d2,
                 Lin l0, Lin l1, float* hid, float* out, bool lastRelu) {
    gemm(s, false, false, x, l0.w, l0.b, hid, M, d1, d0, d0, d1, d1, FLAG_RELU);
    gemm(s, false, false, hid, l1.w, l1.b, out, M, d2, d1, d1, d2, d2, lastRelu ? FLAG_RELU : 0);
}

// GRU update of h[M,H] with input a[M,H] using shared temps zp/rp/hp
static void gru_step(hipStream_t s, Gru g, const float* a, float* h, int M,
                     float* zp, float* rp, float* hp) {
    size_t n = (size_t)M * H;
    int eb = (int)((n + 255) / 256);
    gemm(s, false, false, a, g.w3.w, g.w3.b, zp, M, H, H, H, H, H, 0);
    gemm(s, false, false, h, g.u3.w, g.u3.b, zp, M, H, H, H, H, H, FLAG_ACC);
    gemm(s, false, false, a, g.w4.w, g.w4.b, rp, M, H, H, H, H, H, 0);
    gemm(s, false, false, h, g.u4.w, g.u4.b, rp, M, H, H, H, H, H, FLAG_ACC);
    k_gru_rh<<<eb, 256, 0, s>>>(rp, h, n);
    gemm(s, false, false, a, g.w5.w, g.w5.b, hp, M, H, H, H, H, H, 0);
    gemm(s, false, false, rp, g.u5.w, g.u5.b, hp, M, H, H, H, H, H, FLAG_ACC);
    k_gru_out<<<eb, 256, 0, s>>>(zp, hp, h, n);
}

extern "C" void kernel_launch(void* const* d_in, const int* in_sizes, int n_in,
                              void* d_out, int out_size, void* d_ws, size_t ws_size,
                              hipStream_t stream) {
    (void)in_sizes; (void)n_in; (void)out_size; (void)ws_size;

    const float* roi      = (const float*)d_in[0];   // [NOBJ,H]
    const float* uni      = (const float*)d_in[1];   // [NREL,H]
    const float* emb_ent  = (const float*)d_in[2];   // [NE,DEMB]
    const float* emb_pred = (const float*)d_in[3];   // [NP,DEMB]
    const float* adj_e2e  = (const float*)d_in[4];   // [E,NE,NE]
    const float* adj_e2p  = (const float*)d_in[5];   // [E,NE,NP]
    const float* adj_p2e  = (const float*)d_in[6];   // [E,NP,NE]
    const float* adj_p2p  = (const float*)d_in[7];   // [E,NP,NP]
    const float* P        = (const float*)d_in[8];   // params, jax tree-flatten order
    const int*   labels   = (const int*)d_in[9];
    const int*   sub      = (const int*)d_in[10];
    const int*   obj      = (const int*)d_in[11];
    float*       out      = (float*)d_out;           // ent_logits[2048*151] ++ pred_logits[16384*51]

    // ---- walk params: sorted dict keys at each level, "b" before "w" ----
    size_t po = 0;
    auto lin = [&](int din, int dout) {
        Lin l; l.b = P + po; po += dout; l.w = P + po; po += (size_t)din * dout; return l;
    };
    auto gruP = [&]() {
        Gru g; g.u3 = lin(H, H); g.u4 = lin(H, H); g.u5 = lin(H, H);
        g.w3 = lin(H, H); g.w4 = lin(H, H); g.w5 = lin(H, H); return g;
    };
    Gru gru_ie = gruP(), gru_ip = gruP(), gru_oe = gruP(), gru_op = gruP();
    Lin init_ent  = lin(DEMB, H), init_pred = lin(DEMB, H);
    Lin out_ie0 = lin(H, H), out_ie1 = lin(H, H);
    Lin out_ip0 = lin(H, H), out_ip1 = lin(H, H);
    Lin out_oe0 = lin(H, H), out_oe1 = lin(H, H);
    Lin out_op0 = lin(H, H), out_op1 = lin(H, H);
    Lin recv_ie0 = lin(CAT_I, CAT_I), recv_ie1 = lin(CAT_I, H);
    Lin recv_ip0 = lin(CAT_I, CAT_I), recv_ip1 = lin(CAT_I, H);
    Lin recv_oe0 = lin(CAT_O, CAT_O), recv_oe1 = lin(CAT_O, H);
    Lin recv_op0 = lin(CAT_O, CAT_O), recv_op1 = lin(CAT_O, H);
    Lin send_ie0 = lin(H, HMID), send_ie1 = lin(HMID, Q);
    Lin send_ip0 = lin(H, HMID), send_ip1 = lin(HMID, Q);
    Lin send_oe0 = lin(H, HMID), send_oe1 = lin(HMID, Q);
    Lin send_op0 = lin(H, HMID), send_op1 = lin(HMID, Q);

    // ---- workspace bump allocator ----
    float* W = (float*)d_ws;
    size_t wo = 0;
    auto alloc = [&](size_t n) { float* p = W + wo; wo += n; return p; };

    float* hoe = alloc((size_t)NE * H);
    float* hop = alloc((size_t)NP * H);
    float* hie = alloc((size_t)NOBJ * H);
    float* hip = alloc((size_t)NREL * H);
    float* moe = alloc((size_t)NE * Q);
    float* mop = alloc((size_t)NP * Q);
    float* mie = alloc((size_t)NOBJ * Q);
    float* mip = alloc((size_t)NREL * Q);
    float* tmp = alloc((size_t)NREL * HMID);       // send-MLP hidden (largest), reused
    float* cat_oe = alloc((size_t)NE * CAT_O);
    float* cat_op = alloc((size_t)NP * CAT_O);
    float* cat_ie = alloc((size_t)NOBJ * CAT_I);
    float* cat_ip = alloc((size_t)NREL * CAT_I);
    float* hid    = alloc((size_t)NREL * CAT_I);   // recv-MLP hidden (largest), reused
    float* a_oe = alloc((size_t)NE * H);
    float* a_op = alloc((size_t)NP * H);
    float* a_ie = alloc((size_t)NOBJ * H);
    float* a_ip = alloc((size_t)NREL * H);
    float* zp = alloc((size_t)NREL * H);           // GRU temps, shared by all 4 node types
    float* rp = alloc((size_t)NREL * H);
    float* hp = alloc((size_t)NREL * H);
    float* tout = alloc((size_t)NREL * H);         // out-MLP hidden, reused
    float* f_oe = alloc((size_t)NE * H);
    float* f_op = alloc((size_t)NP * H);
    float* f_ie = alloc((size_t)NOBJ * H);
    float* f_ip = alloc((size_t)NREL * H);
    float* i2o_ent  = alloc((size_t)NOBJ * NE);
    float* i2o_pred = alloc((size_t)NREL * NP);

    float* ent_logits  = out;                          // [NOBJ, NE]
    float* pred_logits = out + (size_t)NOBJ * NE;      // [NREL, NP]

    // ---- init ----
    k_obj_probs<<<((size_t)NOBJ * NE + 255) / 256, 256, 0, stream>>>(labels, i2o_ent);
    zero(stream, i2o_pred, (size_t)NREL * NP);
    gemm(stream, false, false, emb_ent,  init_ent.w,  init_ent.b,  hoe, NE, H, DEMB, DEMB, H, H, 0);
    gemm(stream, false, false, emb_pred, init_pred.w, init_pred.b, hop, NP, H, DEMB, DEMB, H, H, 0);
    hipMemcpyAsync(hie, roi, (size_t)NOBJ * H * sizeof(float), hipMemcpyDeviceToDevice, stream);
    hipMemcpyAsync(hip, uni, (size_t)NREL * H * sizeof(float), hipMemcpyDeviceToDevice, stream);

    for (int t = 0; t < TSTEP; ++t) {
        // ---- send MLPs: H -> 512 -> 256, relu both ----
        mlp2(stream, hoe, NE,   H, HMID, Q, send_oe0, send_oe1, tmp, moe, true);
        mlp2(stream, hop, NP,   H, HMID, Q, send_op0, send_op1, tmp, mop, true);
        mlp2(stream, hie, NOBJ, H, HMID, Q, send_ie0, send_ie1, tmp, mie, true);
        mlp2(stream, hip, NREL, H, HMID, Q, send_ip0, send_ip1, tmp, mip, true);

        // ---- ontology entity: cat -> recv -> GRU ----
        for (int e = 0; e < NEDGE; ++e)
            gemm(stream, true, false, adj_e2e + (size_t)e * NE * NE, moe, nullptr,
                 cat_oe + e * Q, NE, Q, NE, NE, Q, CAT_O, 0);
        for (int e = 0; e < NEDGE; ++e)
            gemm(stream, true, false, adj_p2e + (size_t)e * NP * NE, mop, nullptr,
                 cat_oe + 3 * Q + e * Q, NE, Q, NP, NE, Q, CAT_O, 0);
        gemm(stream, true, false, i2o_ent, mie, nullptr, cat_oe + 6 * Q,
             NE, Q, NOBJ, NE, Q, CAT_O, 0);
        mlp2(stream, cat_oe, NE, CAT_O, CAT_O, H, recv_oe0, recv_oe1, hid, a_oe, true);
        gru_step(stream, gru_oe, a_oe, hoe, NE, zp, rp, hp);

        // ---- ontology predicate ----
        for (int e = 0; e < NEDGE; ++e)
            gemm(stream, true, false, adj_e2p + (size_t)e * NE * NP, moe, nullptr,
                 cat_op + e * Q, NP, Q, NE, NP, Q, CAT_O, 0);
        for (int e = 0; e < NEDGE; ++e)
            gemm(stream, true, false, adj_p2p + (size_t)e * NP * NP, mop, nullptr,
                 cat_op + 3 * Q + e * Q, NP, Q, NP, NP, Q, CAT_O, 0);
        gemm(stream, true, false, i2o_pred, mip, nullptr, cat_op + 6 * Q,
             NP, Q, NREL, NP, Q, CAT_O, 0);
        mlp2(stream, cat_op, NP, CAT_O, CAT_O, H, recv_op0, recv_op1, hid, a_op, true);
        gru_step(stream, gru_op, a_op, hop, NP, zp, rp, hp);

        // ---- image entity: segment sums + i2o_ent@m_oe ----
        zero(stream, cat_ie, (size_t)NOBJ * CAT_I);
        k_segadd<<<((size_t)NREL * Q + 255) / 256, 256, 0, stream>>>(mip, sub, cat_ie, 0);
        k_segadd<<<((size_t)NREL * Q + 255) / 256, 256, 0, stream>>>(mip, obj, cat_ie, Q);
        gemm(stream, false, false, i2o_ent, moe, nullptr, cat_ie + 2 * Q,
             NOBJ, Q, NE, NE, Q, CAT_I, 0);
        mlp2(stream, cat_ie, NOBJ, CAT_I, CAT_I, H, recv_ie0, recv_ie1, hid, a_ie, true);
        gru_step(stream, gru_ie, a_ie, hie, NOBJ, zp, rp, hp);

        // ---- image predicate: gathers + i2o_pred@m_op ----
        k_gather2<<<((size_t)NREL * 512 + 255) / 256, 256, 0, stream>>>(mie, sub, obj, cat_ip);
        gemm(stream, false, false, i2o_pred, mop, nullptr, cat_ip + 2 * Q,
             NREL, Q, NP, NP, Q, CAT_I, 0);
        mlp2(stream, cat_ip, NREL, CAT_I, CAT_I, H, recv_ip0, recv_ip1, hid, a_ip, true);
        gru_step(stream, gru_ip, a_ip, hip, NREL, zp, rp, hp);

        // ---- output heads + logits + softmax feedback ----
        mlp2(stream, hip, NREL, H, H, H, out_ip0, out_ip1, tout, f_ip, false);
        mlp2(stream, hop, NP,   H, H, H, out_op0, out_op1, tout, f_op, false);
        gemm(stream, false, true, f_ip, f_op, nullptr, pred_logits,
             NREL, NP, H, H, H, NP, 0);
        k_softmax_rows<<<(NREL + 63) / 64, 64, 0, stream>>>(pred_logits, i2o_pred, NREL, NP);

        mlp2(stream, hie, NOBJ, H, H, H, out_ie0, out_ie1, tout, f_ie, false);
        mlp2(stream, hoe, NE,   H, H, H, out_oe0, out_oe1, tout, f_oe, false);
        gemm(stream, false, true, f_ie, f_oe, nullptr, ent_logits,
             NOBJ, NE, H, H, H, NE, 0);
        k_softmax_rows<<<(NOBJ + 63) / 64, 64, 0, stream>>>(ent_logits, i2o_ent, NOBJ, NE);
    }
}
